// AgentDRQN_72258529787985
// MI455X (gfx1250) — compile-verified, matrix operand order checked by reading, not run on hardware
//
#include <hip/hip_runtime.h>
#include <hip/hip_bf16.h>

// ---------------------------------------------------------------------------
// Problem constants (from reference)
// ---------------------------------------------------------------------------
#define BB   32
#define LL   64
#define NN   8
#define NIMG (BB * LL * NN)   // 16384, flat (b,l,n) order
#define HID  64
#define NACT 8

typedef __attribute__((ext_vector_type(16))) __bf16 v16bf;
typedef __attribute__((ext_vector_type(8)))  float  v8f;
typedef __attribute__((ext_vector_type(4)))  int    v4i;

#define WMMA_BF16(a, b, c) \
  __builtin_amdgcn_wmma_f32_16x16x32_bf16(false, (a), false, (b), (short)0, (c), false, false)

// gfx1250 async global->LDS copy (ASYNCcnt path), guarded: fall back to plain
// global loads if this toolchain does not declare the builtin.
// Builtin signature (from hipcc diagnostic): (v4i AS1*, v4i AS3*, imm int, imm int)
#if __has_builtin(__builtin_amdgcn_global_load_async_to_lds_b128)
#define HAVE_ASYNC_LDS 1
typedef __attribute__((address_space(1))) v4i* g_v4i_p;
typedef __attribute__((address_space(3))) v4i* l_v4i_p;
__device__ __forceinline__ void async_copy_b128(const void* gsrc, void* ldst) {
  __builtin_amdgcn_global_load_async_to_lds_b128(
      (g_v4i_p)(uintptr_t)gsrc,
      (l_v4i_p)(uint32_t)(uintptr_t)ldst,   // low 32 bits of flat LDS addr = LDS offset
      0, 0);
}
__device__ __forceinline__ void async_wait_all() {
#if __has_builtin(__builtin_amdgcn_s_wait_asynccnt)
  __builtin_amdgcn_s_wait_asynccnt(0);
#else
  asm volatile("s_wait_asynccnt 0x0" ::: "memory");
#endif
}
#endif

// ---------------------------------------------------------------------------
// Fragment loader (CDNA5 16-bit A/B 16x32 layout, CDNA5 ISA 7.12.2)
// lane -> row (M for A, N for B) = lane&15 ; kbase = (lane<16)?0:8
// VGPR v in [0,3]: K = kbase + 2v, +1 ; VGPR v in [4,7]: K = 16 + kbase + 2(v-4), +1
// (8 consecutive bf16 per half -> compiler emits two b128 loads per fragment)
// ---------------------------------------------------------------------------
__device__ __forceinline__ v16bf load_frag_bf16(const __bf16* rowptr, int lane) {
  const int kb = (lane < 16) ? 0 : 8;
  v16bf f;
#pragma unroll
  for (int v = 0; v < 8; ++v) {
    const int k0 = ((v < 4) ? (2 * v) : (16 + 2 * (v - 4))) + kb;
    f[2 * v]     = rowptr[k0];
    f[2 * v + 1] = rowptr[k0 + 1];
  }
  return f;
}

// ---------------------------------------------------------------------------
// Weight fp32 -> bf16 conversion (fc_w padded with zero rows)
// ---------------------------------------------------------------------------
__global__ void cvt_bf16_kernel(const float* __restrict__ src, __bf16* __restrict__ dst,
                                int nsrc, int ndst) {
  int i = blockIdx.x * blockDim.x + threadIdx.x;
  if (i < ndst) dst[i] = (__bf16)((i < nsrc) ? src[i] : 0.0f);
}

// ---------------------------------------------------------------------------
// Fused per-image CNN: conv1(8x8 s4) + relu -> conv2(4x4 s2) + relu ->
// conv3(3x3) + relu, all intermediates in LDS. One block = one image.
// ---------------------------------------------------------------------------
__global__ __launch_bounds__(128) void cnn_fused_kernel(
    const float* __restrict__ obs,     // (NIMG, 3, 40, 40)
    const __bf16* __restrict__ w1,     // (32, 192) bf16
    const float* __restrict__ b1,      // (32,)
    const __bf16* __restrict__ w2,     // (64, 512) bf16
    const float* __restrict__ b2,      // (64,)
    const __bf16* __restrict__ w3,     // (64, 576) bf16
    const float* __restrict__ b3,      // (64,)
    __bf16* __restrict__ feats)        // (NIMG, 64) bf16
{
  __shared__ __bf16 Bsm[96 * 192];     // conv1 im2col (n-major [p][k]); reused for conv2
  __shared__ __bf16 C1[32 * 81];       // conv1 output [cout][p]
  __shared__ __bf16 C2[64 * 9];        // conv2 output [cout*9 + p]
#if HAVE_ASYNC_LDS
  __shared__ __align__(16) float ImgSm[4800];  // raw image staged via async DMA
#endif

  const int img  = blockIdx.x;
  const int lane = threadIdx.x & 31;
  const int wv   = threadIdx.x >> 5;
  const int mrow = lane & 15;
  const int rbase = (lane >= 16) ? 8 : 0;
  const float* x = obs + (size_t)img * 4800;

#if HAVE_ASYNC_LDS
  // ---- stage raw image into LDS with async global->LDS b128 copies ----
  for (int c = threadIdx.x; c < 1200; c += 128)      // 1200 x 16B = 19200B
    async_copy_b128(x + c * 4, ImgSm + c * 4);
  async_wait_all();
  __syncthreads();
#endif

  // ---- conv1 im2col: K = c*64 + ky*8 + kx (matches OIHW weight flatten) ----
  for (int i = threadIdx.x; i < 96 * 192; i += 128) {
    const int p = i / 192, k = i - p * 192;
    float v = 0.0f;
    if (p < 81) {
      const int oy = p / 9, ox = p - oy * 9;
      const int c = k >> 6, rem = k & 63, ky = rem >> 3, kx = rem & 7;
#if HAVE_ASYNC_LDS
      v = ImgSm[c * 1600 + (oy * 4 + ky) * 40 + (ox * 4 + kx)];
#else
      v = x[c * 1600 + (oy * 4 + ky) * 40 + (ox * 4 + kx)];
#endif
    }
    Bsm[i] = (__bf16)v;
  }
  __syncthreads();

  // ---- conv1: M=32 (2 tiles) x N=96 (6 tiles), K=192 (6 wmma steps) ----
  for (int tid = wv; tid < 12; tid += 4) {
    const int tm = tid / 6, tn = tid - tm * 6;
    const __bf16* aptr = w1 + (tm * 16 + mrow) * 192;
    const __bf16* bptr = &Bsm[(tn * 16 + mrow) * 192];
    v8f acc = {0.f, 0.f, 0.f, 0.f, 0.f, 0.f, 0.f, 0.f};
#pragma unroll
    for (int ks = 0; ks < 6; ++ks) {
      v16bf a = load_frag_bf16(aptr + ks * 32, lane);
      v16bf b = load_frag_bf16(bptr + ks * 32, lane);
      acc = WMMA_BF16(a, b, acc);
    }
    const int p = tn * 16 + mrow;           // D: lane -> column (position)
    if (p < 81) {
#pragma unroll
      for (int r = 0; r < 8; ++r) {
        const int cout = tm * 16 + rbase + r;
        float y = acc[r] + b1[cout];
        C1[cout * 81 + p] = (__bf16)(y > 0.0f ? y : 0.0f);
      }
    }
  }
  __syncthreads();

  // ---- conv2 im2col into reused Bsm: K = cin*16 + ky*4 + kx ----
  for (int i = threadIdx.x; i < 16 * 512; i += 128) {
    const int p = i >> 9, k = i & 511;
    float v = 0.0f;
    if (p < 9) {
      const int oy = p / 3, ox = p - oy * 3;
      const int cin = k >> 4, rem = k & 15, ky = rem >> 2, kx = rem & 3;
      v = (float)C1[cin * 81 + (oy * 2 + ky) * 9 + (ox * 2 + kx)];
    }
    Bsm[i] = (__bf16)v;
  }
  __syncthreads();

  // ---- conv2: M=64 (4 tiles, one per wave) x N=16 (9 valid), K=512 ----
  {
    const int tm = wv;
    const __bf16* aptr = w2 + (tm * 16 + mrow) * 512;
    const __bf16* bptr = &Bsm[mrow * 512];
    v8f acc = {0.f, 0.f, 0.f, 0.f, 0.f, 0.f, 0.f, 0.f};
#pragma unroll
    for (int ks = 0; ks < 16; ++ks) {
      v16bf a = load_frag_bf16(aptr + ks * 32, lane);
      v16bf b = load_frag_bf16(bptr + ks * 32, lane);
      acc = WMMA_BF16(a, b, acc);
    }
    const int p = mrow;
    if (p < 9) {
#pragma unroll
      for (int r = 0; r < 8; ++r) {
        const int cout = tm * 16 + rbase + r;
        float y = acc[r] + b2[cout];
        C2[cout * 9 + p] = (__bf16)(y > 0.0f ? y : 0.0f);
      }
    }
  }
  __syncthreads();

  // ---- conv3: GEMV via wmma, B column broadcast (K = cin*9 + p = C2 layout) ----
  {
    const int tm = wv;
    const __bf16* aptr = w3 + (tm * 16 + mrow) * 576;
    v8f acc = {0.f, 0.f, 0.f, 0.f, 0.f, 0.f, 0.f, 0.f};
#pragma unroll
    for (int ks = 0; ks < 18; ++ks) {
      v16bf a = load_frag_bf16(aptr + ks * 32, lane);
      v16bf b = load_frag_bf16(&C2[0] + ks * 32, lane);  // all columns identical
      acc = WMMA_BF16(a, b, acc);
    }
    if (mrow == 0) {                       // column 0 holds the result
#pragma unroll
      for (int r = 0; r < 8; ++r) {
        const int cout = tm * 16 + rbase + r;
        float y = acc[r] + b3[cout];
        feats[(size_t)img * 64 + cout] = (__bf16)(y > 0.0f ? y : 0.0f);
      }
    }
  }
}

// ---------------------------------------------------------------------------
// GRU: rows (b,n) independent across time; each block owns 16 rows for all 64
// steps. gi = x @ W_ih^T, gh = h @ W_hh^T via wmma (M=16, N=192, K=64).
// Hidden kept in f32 + bf16 shadow (so gh A-fragments are b128 DS loads).
// ---------------------------------------------------------------------------
__global__ __launch_bounds__(128) void gru_kernel(
    const __bf16* __restrict__ feats,  // (NIMG, 64) bf16, img = ((b*L+l)*N+n)
    const float* __restrict__ h0,      // (B*N, 64)
    const __bf16* __restrict__ wih,    // (192, 64) bf16
    const __bf16* __restrict__ whh,    // (192, 64) bf16
    const float* __restrict__ bih,     // (192,)
    const float* __restrict__ bhh,     // (192,)
    __bf16* __restrict__ hs,           // (NIMG, 64) bf16, same img order
    float* __restrict__ hlast)         // (B*N, 64) f32 -> d_out tail
{
  __shared__ float  Hs[16 * 64];
  __shared__ __bf16 Hsb[16 * 64];
  __shared__ float  Gi[16 * 192];
  __shared__ float  Gh[16 * 192];

  const int row0 = blockIdx.x * 16;
  const int lane = threadIdx.x & 31;
  const int wv   = threadIdx.x >> 5;
  const int mrow = lane & 15;
  const int rbase = (lane >= 16) ? 8 : 0;

  for (int i = threadIdx.x; i < 16 * 64; i += 128) {
    const float h = h0[(size_t)(row0 + (i >> 6)) * 64 + (i & 63)];
    Hs[i]  = h;
    Hsb[i] = (__bf16)h;
  }

  // per-lane row -> (b, n) for feats/hs addressing
  const int r_lane = row0 + mrow;
  const int b_lane = r_lane >> 3;
  const int n_lane = r_lane & 7;

  for (int t = 0; t < 64; ++t) {
    __syncthreads();  // Hs/Hsb/Gi/Gh from previous step fully consumed & visible

    for (int job = wv; job < 24; job += 4) {      // wave-uniform
      const int g = (job < 12) ? job : (job - 12);
      const bool is_gh = (job >= 12);
      const __bf16* aptr = is_gh ? (&Hsb[mrow * 64])
                                 : (feats + (size_t)((b_lane * LL + t) * NN + n_lane) * 64);
      const __bf16* bptr = (is_gh ? whh : wih) + (g * 16 + mrow) * 64;
      v8f acc = {0.f, 0.f, 0.f, 0.f, 0.f, 0.f, 0.f, 0.f};
#pragma unroll
      for (int ks = 0; ks < 2; ++ks) {
        v16bf a = load_frag_bf16(aptr + ks * 32, lane);
        v16bf b = load_frag_bf16(bptr + ks * 32, lane);
        acc = WMMA_BF16(a, b, acc);
      }
      float* G = is_gh ? Gh : Gi;
#pragma unroll
      for (int r = 0; r < 8; ++r)
        G[(rbase + r) * 192 + g * 16 + mrow] = acc[r];
    }
    __syncthreads();

    // elementwise gate math; each (m, j) owned by exactly one thread
    for (int i = threadIdx.x; i < 16 * 64; i += 128) {
      const int m = i >> 6, j = i & 63;
      const float gr = Gi[m * 192 + j]       + bih[j]       + Gh[m * 192 + j]       + bhh[j];
      const float gz = Gi[m * 192 + 64 + j]  + bih[64 + j]  + Gh[m * 192 + 64 + j]  + bhh[64 + j];
      const float gn = Gi[m * 192 + 128 + j] + bih[128 + j];
      const float hn_pre = Gh[m * 192 + 128 + j] + bhh[128 + j];
      const float rg = 1.0f / (1.0f + __expf(-gr));
      const float zg = 1.0f / (1.0f + __expf(-gz));
      const float nng = tanhf(gn + rg * hn_pre);
      const float h = Hs[i];
      const float hnew = (1.0f - zg) * nng + zg * h;
      Hs[i]  = hnew;
      Hsb[i] = (__bf16)hnew;
      const int rr = row0 + m, bb2 = rr >> 3, nn2 = rr & 7;
      hs[(size_t)((bb2 * LL + t) * NN + nn2) * 64 + j] = (__bf16)hnew;
    }
  }
  __syncthreads();

  for (int i = threadIdx.x; i < 16 * 64; i += 128)
    hlast[(size_t)(row0 + (i >> 6)) * 64 + (i & 63)] = Hs[i];
}

// ---------------------------------------------------------------------------
// FC: q = rnn_out @ fc_w^T + fc_b ; one wmma tile (16 rows x 16 cols, 8 valid)
// per wave, K=64.
// ---------------------------------------------------------------------------
__global__ __launch_bounds__(128) void fc_kernel(
    const __bf16* __restrict__ hs,    // (NIMG, 64) bf16 in (b,l,n) row order
    const __bf16* __restrict__ wfc,   // (16, 64) bf16, rows 8..15 zero
    const float* __restrict__ bfc,    // (8,)
    float* __restrict__ q)            // (NIMG, 8) -> d_out head
{
  const int wgid = blockIdx.x * 4 + (threadIdx.x >> 5);
  const int img0 = wgid * 16;
  const int lane = threadIdx.x & 31;
  const int mrow = lane & 15;
  const int rbase = (lane >= 16) ? 8 : 0;

  const __bf16* aptr = hs + (size_t)(img0 + mrow) * 64;
  const __bf16* bptr = wfc + mrow * 64;
  v8f acc = {0.f, 0.f, 0.f, 0.f, 0.f, 0.f, 0.f, 0.f};
#pragma unroll
  for (int ks = 0; ks < 2; ++ks) {
    v16bf a = load_frag_bf16(aptr + ks * 32, lane);
    v16bf b = load_frag_bf16(bptr + ks * 32, lane);
    acc = WMMA_BF16(a, b, acc);
  }
  if (mrow < NACT) {
#pragma unroll
    for (int r = 0; r < 8; ++r)
      q[(size_t)(img0 + rbase + r) * NACT + mrow] = acc[r] + bfc[mrow];
  }
}

// ---------------------------------------------------------------------------
// Host launcher
// ---------------------------------------------------------------------------
extern "C" void kernel_launch(void* const* d_in, const int* in_sizes, int n_in,
                              void* d_out, int out_size, void* d_ws, size_t ws_size,
                              hipStream_t stream) {
  (void)in_sizes; (void)n_in; (void)out_size; (void)ws_size;

  const float* obs  = (const float*)d_in[0];   // (32,64,8,3,40,40)
  const float* h0   = (const float*)d_in[1];   // (32,8,64)
  const float* w1f  = (const float*)d_in[2];   // (32,3,8,8)
  const float* b1   = (const float*)d_in[3];
  const float* w2f  = (const float*)d_in[4];   // (64,32,4,4)
  const float* b2   = (const float*)d_in[5];
  const float* w3f  = (const float*)d_in[6];   // (64,64,3,3)
  const float* b3   = (const float*)d_in[7];
  const float* wihf = (const float*)d_in[8];   // (192,64)
  const float* whhf = (const float*)d_in[9];   // (192,64)
  const float* bih  = (const float*)d_in[10];
  const float* bhh  = (const float*)d_in[11];
  const float* wfcf = (const float*)d_in[12];  // (8,64)
  const float* bfc  = (const float*)d_in[13];

  float* q_out  = (float*)d_out;                       // (16384, 8)
  float* h_out  = (float*)d_out + (size_t)NIMG * NACT; // (256, 64)

  // workspace carve-up (256B aligned)
  char* ws = (char*)d_ws;
  size_t off = 0;
  auto carve = [&](size_t bytes) -> void* {
    void* p = ws + off;
    off = (off + bytes + 255) & ~(size_t)255;
    return p;
  };
  __bf16* w1b  = (__bf16*)carve(6144u * 2);
  __bf16* w2b  = (__bf16*)carve(32768u * 2);
  __bf16* w3b  = (__bf16*)carve(36864u * 2);
  __bf16* wihb = (__bf16*)carve(12288u * 2);
  __bf16* whhb = (__bf16*)carve(12288u * 2);
  __bf16* wfcb = (__bf16*)carve(1024u * 2);
  __bf16* feats = (__bf16*)carve((size_t)NIMG * 64 * 2);
  __bf16* hsbuf = (__bf16*)carve((size_t)NIMG * 64 * 2);

  auto cvt = [&](const float* s, __bf16* d, int nsrc, int ndst) {
    cvt_bf16_kernel<<<(ndst + 255) / 256, 256, 0, stream>>>(s, d, nsrc, ndst);
  };
  cvt(w1f,  w1b,  6144, 6144);
  cvt(w2f,  w2b,  32768, 32768);
  cvt(w3f,  w3b,  36864, 36864);
  cvt(wihf, wihb, 12288, 12288);
  cvt(whhf, whhb, 12288, 12288);
  cvt(wfcf, wfcb, 512, 1024);          // pad fc_w to 16 rows with zeros

  cnn_fused_kernel<<<NIMG, 128, 0, stream>>>(obs, w1b, b1, w2b, b2, w3b, b3, feats);
  gru_kernel<<<(BB * NN) / 16, 128, 0, stream>>>(feats, h0, wihb, whhb, bih, bhh, hsbuf, h_out);
  fc_kernel<<<NIMG / 16 / 4, 128, 0, stream>>>(hsbuf, wfcb, bfc, q_out);
}
